// LstmEncoder_76922864271611
// MI455X (gfx1250) — matrix operand order repaired and checked by reading, not compiled, not run on hardware
//
#include <hip/hip_runtime.h>
#include <hip/hip_bf16.h>

// Problem constants (match reference)
#define B_   64
#define T_   512
#define H_   512
#define E_   256
#define M_   (T_ * B_)   // 32768 rows (t*B + b)
#define G4H  2048        // 4*H gate width

typedef __attribute__((ext_vector_type(16))) __bf16 v16bf;
typedef __attribute__((ext_vector_type(8)))  float  v8f;

union FragU { v16bf v; uint4 q[2]; };

#if defined(__has_builtin)
#if __has_builtin(__builtin_amdgcn_tensor_load_to_lds)
#define HAVE_TDM 1
#endif
#endif

#ifdef HAVE_TDM
typedef __attribute__((ext_vector_type(4))) unsigned tdm_g0_t;
typedef __attribute__((ext_vector_type(8))) int      tdm_g1_t;
typedef __attribute__((ext_vector_type(4))) int      tdm_g2_t;
#endif

__device__ __forceinline__ unsigned short f2bf(float f) {
  unsigned u = __float_as_uint(f);
  unsigned r = 0x7FFFu + ((u >> 16) & 1u);   // round-to-nearest-even
  return (unsigned short)((u + r) >> 16);
}

__device__ __forceinline__ v8f wmma_bf16(v16bf a, v16bf b, v8f c) {
  // D = A(16x32 bf16) * B(32x16 bf16) + C(16x16 f32)
  return __builtin_amdgcn_wmma_f32_16x16x32_bf16(false, a, false, b, (short)0, c,
                                                 false, false);
}

// A fragment (16x32, bf16) from row-major [*, lda] at tile (m0,k0).
// Layout: lanes 0-15 -> M=lane, K=0..15;
//         lanes 16-31 -> M=lane-16, K=16..31 (matches load_b).
__device__ __forceinline__ v16bf load_a(const unsigned short* base, int lda,
                                        int m0, int k0) {
  int lane = threadIdx.x & 31;
  int row = lane & 15, hi = lane >> 4;
  const unsigned short* p = base + (size_t)(m0 + row) * lda + k0 + hi * 16;
  FragU u;
  u.q[0] = *reinterpret_cast<const uint4*>(p);        // K = hi*16 + 0..7
  u.q[1] = *reinterpret_cast<const uint4*>(p + 8);    // K = hi*16 + 8..15
  return u.v;
}

// B fragment (32x16, bf16) where B[k][n] = W[n][k], W row-major [N][ldb].
// lanes 0-15: N=lane, K=0..15 ; lanes 16-31: N=lane-16, K=16..31 => 32B contiguous/lane.
__device__ __forceinline__ v16bf load_b(const unsigned short* W, int ldb,
                                        int n0, int k0) {
  int lane = threadIdx.x & 31;
  int col = lane & 15, hi = lane >> 4;
  const uint4* p = reinterpret_cast<const uint4*>(W + (size_t)(n0 + col) * ldb +
                                                  k0 + hi * 16);
  FragU u;
  u.q[0] = p[0];
  u.q[1] = p[1];
  return u.v;
}

__device__ __forceinline__ float sigm(float x) { return 1.0f / (1.0f + __expf(-x)); }
__device__ __forceinline__ float tanh_f(float x) {
  float e = __expf(-2.0f * fabsf(x));
  float r = (1.0f - e) / (1.0f + e);
  return copysignf(r, x);
}

// ---------------------------------------------------------------------------
// Embedding gather: x[t*B+b][d] = bf16(table[words[b][t]][d])
__global__ void embed_kernel(const int* __restrict__ words,
                             const float* __restrict__ table,
                             unsigned short* __restrict__ x) {
  int row = blockIdx.x;          // t*B + b
  int t = row / B_, b = row % B_;
  int w = words[(size_t)b * T_ + t];
  if (w < 0) w = 32000 - 1;      // pad_index == -1 -> n_words-1
  x[(size_t)row * E_ + threadIdx.x] =
      f2bf(table[(size_t)w * E_ + threadIdx.x]);
}

// f32 -> bf16 bulk convert (weights)
__global__ void cvt_bf16_kernel(const float* __restrict__ src,
                                unsigned short* __restrict__ dst, int n) {
  for (int i = blockIdx.x * blockDim.x + threadIdx.x; i < n;
       i += gridDim.x * blockDim.x)
    dst[i] = f2bf(src[i]);
}

// ---------------------------------------------------------------------------
// Input GEMM: xg[dir][m][n] = sum_k x[m][k] * wih[dir][n][k] + bias[dir][n]
// 256 thr = 8 waves as 4(M)x2(N); wave tile 32x64 (2x4 of 16x16); block 128x128.
// 8 WMMAs per 6 fragment loads per K-chunk.
__global__ __launch_bounds__(256) void input_gemm_kernel(
    const unsigned short* __restrict__ x, int K,
    const unsigned short* __restrict__ wih,   // [2][2048][K] bf16
    const float* __restrict__ bias,           // [2][2048]
    float* __restrict__ xg) {                 // [2][M_][2048]
  const int dir = blockIdx.z;
  const int wave = threadIdx.x >> 5;
  const int wm = wave & 3, wn = wave >> 2;
  const int m0 = blockIdx.x * 128 + wm * 32;
  const int n0 = blockIdx.y * 128 + wn * 64;
  const unsigned short* W = wih + (size_t)dir * G4H * K;

  v8f acc[2][4] = {};
  for (int k = 0; k < K; k += 32) {
    if (k + 32 < K) {  // hint next A chunk into cache
      __builtin_prefetch(x + (size_t)m0 * K + k + 32, 0, 1);
    }
    v16bf a0 = load_a(x, K, m0, k);
    v16bf a1 = load_a(x, K, m0 + 16, k);
#pragma unroll
    for (int ni = 0; ni < 4; ++ni) {
      v16bf b = load_b(W, K, n0 + ni * 16, k);
      acc[0][ni] = wmma_bf16(a0, b, acc[0][ni]);
      acc[1][ni] = wmma_bf16(a1, b, acc[1][ni]);
    }
  }

  const int lane = threadIdx.x & 31;
  const int colh = lane & 15, hi = lane >> 4;
#pragma unroll
  for (int mi = 0; mi < 2; ++mi)
#pragma unroll
    for (int ni = 0; ni < 4; ++ni)
#pragma unroll
      for (int j = 0; j < 8; ++j) {
        int row = m0 + mi * 16 + j + 8 * hi;
        int n = n0 + ni * 16 + colh;
        xg[(size_t)dir * M_ * G4H + (size_t)row * G4H + n] =
            acc[mi][ni][j] + bias[(size_t)dir * G4H + n];
      }
}

// ---------------------------------------------------------------------------
__global__ void reset_barrier_kernel(unsigned* bar) {
  if (threadIdx.x == 0) { bar[0] = 0u; bar[1] = 0u; }
}

__device__ __forceinline__ void grid_sync(unsigned* cnt, unsigned* gen,
                                          unsigned total) {
  __syncthreads();
  __threadfence();  // release this block's stores to device scope
  if (threadIdx.x == 0) {
    unsigned g = __hip_atomic_load(gen, __ATOMIC_RELAXED, __HIP_MEMORY_SCOPE_AGENT);
    unsigned arrived =
        __hip_atomic_fetch_add(cnt, 1u, __ATOMIC_ACQ_REL, __HIP_MEMORY_SCOPE_AGENT);
    if (arrived == total - 1u) {
      __hip_atomic_store(cnt, 0u, __ATOMIC_RELAXED, __HIP_MEMORY_SCOPE_AGENT);
      __hip_atomic_store(gen, g + 1u, __ATOMIC_RELEASE, __HIP_MEMORY_SCOPE_AGENT);
    } else {
      while (__hip_atomic_load(gen, __ATOMIC_ACQUIRE, __HIP_MEMORY_SCOPE_AGENT) == g)
        __builtin_amdgcn_s_sleep(8);
    }
  }
  __syncthreads();
  __threadfence();  // acquire other blocks' stores
}

// Persistent bidirectional recurrence. grid = (32 hidden tiles, 2 dirs),
// block = 128 (4 waves over batch tiles of 16). Each block owns a 64x16
// (batch x hidden) strip of h/c and the matching 64-row w_hh strip in LDS.
__global__ __launch_bounds__(128) void lstm_recurrence_kernel(
    const unsigned short* __restrict__ whh,  // [2][2048][512] bf16
    const float* __restrict__ xg,            // [2][M_][2048]
    unsigned short* __restrict__ hbuf,       // [2 dir][2 parity][64][512] bf16
    float* __restrict__ cbuf,                // [2][64][512]
    unsigned short* __restrict__ y_bf,       // [M_][1024] or nullptr
    float* __restrict__ out_f32,             // d_out [B][T][1024] or nullptr
    unsigned* bar) {
  const int dir = blockIdx.y;
  const int nh = blockIdx.x;          // hidden tile 0..31
  const int n0 = nh * 16;
  const int wave = threadIdx.x >> 5;  // batch tile 0..3
  const int m0 = wave * 16;
  const int lane = threadIdx.x & 31;
  const int colh = lane & 15, hi = lane >> 4;

  // Stage w_hh strip: LDS row r = g*16+cr  <-  whh[dir][g*512 + n0 + cr][0..512)
  __shared__ unsigned short lds_w[64 * 512];  // 64 KB
  const unsigned short* wsrc = whh + (size_t)dir * G4H * H_;
#ifdef HAVE_TDM
  // Tensor Data Mover: four 2D tile loads (16 rows x 1024B, stride 1024B),
  // one per gate, issued by wave 0 only (TDM ignores EXEC; one issue each).
  if (wave == 0) {
#pragma unroll
    for (int g = 0; g < 4; ++g) {
      // Generic->LDS address: low 32 bits of a shared-memory pointer are the
      // LDS byte offset.
      unsigned lds_addr = (unsigned)(uintptr_t)(&lds_w[(g * 16) * 512]);
      unsigned long long ga =
          (unsigned long long)(uintptr_t)(wsrc + (size_t)(g * H_ + n0) * H_);
      tdm_g0_t g0;
      g0[0] = 1u;                                   // count=1 (user descriptor)
      g0[1] = lds_addr;                             // lds_addr [63:32]
      g0[2] = (unsigned)(ga & 0xFFFFFFFFu);         // global_addr [95:64]
      g0[3] = (unsigned)((ga >> 32) & 0x01FFFFFFu)  // global_addr [120:96]
              | (2u << 30);                         // type=2 ("image") [127:126]
      tdm_g1_t g1;
      g1[0] = (int)(1u << 16);    // data_size=1 -> 2 bytes  [17:16]
      g1[1] = (int)(512u << 16);  // tensor_dim0[15:0]=512 in [63:48]
      g1[2] = (int)(16u << 16);   // tensor_dim0 hi=0; tensor_dim1[15:0]=16 in [95:80]
      g1[3] = (int)(512u << 16);  // tensor_dim1 hi=0; tile_dim0=512 in [127:112]
      g1[4] = (int)16;            // tile_dim1=16 in [143:128]; tile_dim2=0
      g1[5] = (int)512;           // tensor_dim0_stride low32 = 512 elements
      g1[6] = 0;                  // stride hi; tensor_dim1_stride lo
      g1[7] = 0;
      tdm_g2_t gz = {0, 0, 0, 0};
#if __clang_major__ >= 23
      tdm_g1_t gz8 = {0, 0, 0, 0, 0, 0, 0, 0};
      __builtin_amdgcn_tensor_load_to_lds(g0, g1, gz, gz, gz8, 0);
#else
      __builtin_amdgcn_tensor_load_to_lds(g0, g1, gz, gz, 0);
#endif
    }
    __builtin_amdgcn_s_wait_tensorcnt(0);
  }
#else
  for (int i = threadIdx.x; i < 64 * 512 / 8; i += blockDim.x) {
    int r = i >> 6;          // 64 uint4-chunks per row
    int chunk = i & 63;
    int g = r >> 4, cr = r & 15;
    const uint4* src = reinterpret_cast<const uint4*>(
        wsrc + (size_t)(g * H_ + n0 + cr) * H_ + chunk * 8);
    *reinterpret_cast<uint4*>(&lds_w[r * 512 + chunk * 8]) = *src;
  }
#endif
  __syncthreads();

  const unsigned short* lds_base = lds_w;  // generic pointer -> ds loads

  for (int s = 0; s < T_; ++s) {
    const int t = dir ? (T_ - 1 - s) : s;
    const unsigned short* hprev = hbuf + (size_t)(dir * 2 + (s & 1)) * B_ * H_;
    unsigned short* hnext = hbuf + (size_t)(dir * 2 + ((s + 1) & 1)) * B_ * H_;

    v8f acc[4] = {};
    if (s > 0) {  // step 0: h == 0, skip GEMM entirely
      for (int k = 0; k < H_; k += 32) {
        v16bf a = load_a(hprev, H_, m0, k);
#pragma unroll
        for (int g = 0; g < 4; ++g) {
          v16bf b = load_b(lds_base, 512, g * 16, k);  // LDS-resident weights
          acc[g] = wmma_bf16(a, b, acc[g]);
        }
      }
    }

    // Gates + state update for this wave's 16x16 tile
    const float* xgrow =
        xg + (size_t)dir * M_ * G4H + ((size_t)t * B_) * G4H;
#pragma unroll
    for (int j = 0; j < 8; ++j) {
      int r = m0 + j + 8 * hi;   // batch index
      int n = n0 + colh;         // hidden index
      const float* xr = xgrow + (size_t)r * G4H;
      float gi = acc[0][j] + xr[n];
      float gf = acc[1][j] + xr[H_ + n];
      float gg = acc[2][j] + xr[2 * H_ + n];
      float go = acc[3][j] + xr[3 * H_ + n];
      size_t cidx = ((size_t)dir * B_ + r) * H_ + n;
      float cold = (s == 0) ? 0.0f : cbuf[cidx];
      float cnew = sigm(gf) * cold + sigm(gi) * tanh_f(gg);
      cbuf[cidx] = cnew;
      float hh = sigm(go) * tanh_f(cnew);
      hnext[(size_t)r * H_ + n] = f2bf(hh);
      if (y_bf)  // next layer input [t*B+b][2H] bf16
        y_bf[((size_t)t * B_ + r) * (2 * H_) + dir * H_ + n] = f2bf(hh);
      if (out_f32)  // final output [B][T][2H] f32
        out_f32[((size_t)r * T_ + t) * (2 * H_) + dir * H_ + n] = hh;
    }
    grid_sync(&bar[0], &bar[1], 64u);
  }
}

// ---------------------------------------------------------------------------
extern "C" void kernel_launch(void* const* d_in, const int* in_sizes, int n_in,
                              void* d_out, int out_size, void* d_ws, size_t ws_size,
                              hipStream_t stream) {
  const int* words = (const int*)d_in[0];
  const float* table = (const float*)d_in[1];

  // Workspace carve-up (~701 MB)
  char* p = (char*)d_ws;
  auto take = [&](size_t bytes) -> void* {
    void* r = (void*)p;
    p += (bytes + 255) & ~(size_t)255;
    return r;
  };
  unsigned short* x0 = (unsigned short*)take((size_t)M_ * E_ * 2);           // 16 MB
  unsigned short* y0 = (unsigned short*)take((size_t)M_ * 2 * H_ * 2);       // 64 MB
  unsigned short* y1 = (unsigned short*)take((size_t)M_ * 2 * H_ * 2);       // 64 MB
  float* xg = (float*)take((size_t)2 * M_ * G4H * 4);                        // 512 MB
  unsigned short* wih_bf = (unsigned short*)take((size_t)2 * G4H * 1024 * 2);// 8 MB
  unsigned short* whh_bf = (unsigned short*)take((size_t)2 * G4H * H_ * 2);  // 4 MB
  unsigned short* hbuf = (unsigned short*)take((size_t)2 * 2 * B_ * H_ * 2);
  float* cbuf = (float*)take((size_t)2 * B_ * H_ * 4);
  unsigned* bar = (unsigned*)take(256);

  reset_barrier_kernel<<<1, 32, 0, stream>>>(bar);
  embed_kernel<<<dim3(M_), dim3(E_), 0, stream>>>(words, table, x0);

  const unsigned short* xin = x0;
  int K = E_;
  for (int l = 0; l < 3; ++l) {
    const float* wih = (const float*)d_in[2 + 3 * l];
    const float* whh = (const float*)d_in[3 + 3 * l];
    const float* bias = (const float*)d_in[4 + 3 * l];

    cvt_bf16_kernel<<<dim3(1024), dim3(256), 0, stream>>>(wih, wih_bf, 2 * G4H * K);
    cvt_bf16_kernel<<<dim3(1024), dim3(256), 0, stream>>>(whh, whh_bf, 2 * G4H * H_);

    input_gemm_kernel<<<dim3(M_ / 128, G4H / 128, 2), dim3(256), 0, stream>>>(
        xin, K, wih_bf, bias, xg);

    const bool last = (l == 2);
    unsigned short* ybuf = last ? nullptr : (l == 0 ? y0 : y1);
    float* fout = last ? (float*)d_out : nullptr;
    lstm_recurrence_kernel<<<dim3(32, 2), dim3(128), 0, stream>>>(
        whh_bf, xg, hbuf, cbuf, ybuf, fout, bar);

    xin = (l == 0) ? y0 : y1;
    K = 2 * H_;
  }
}
